// GraphAttentionLayer_120259084551
// MI455X (gfx1250) — matrix operand order repaired and checked by reading, not compiled, not run on hardware
//
#include <hip/hip_runtime.h>
#include <math.h>

typedef __attribute__((ext_vector_type(16))) _Float16 v16h;
typedef __attribute__((ext_vector_type(8)))  float    v8f;
typedef __attribute__((ext_vector_type(4)))  float    f32x4;

#define GAT_B     16
#define GAT_N     2048
#define GAT_F     64
#define GAT_SLOPE 0.2f

// ---------------------------------------------------------------------------
// Kernel 1: Wh = h @ W (f32), store WhT as f16 [B][F][N] (transposed so the
// WMMA B-matrix per-lane K-run is contiguous), plus s1 = Wh.a1, s2 = Wh.a2.
// 64 rows per 256-thread block; W (16KB) and the h tile staged in LDS.
// ---------------------------------------------------------------------------
__global__ __launch_bounds__(256) void gat_prep(const float* __restrict__ h,
                                                const float* __restrict__ W,
                                                const float* __restrict__ a1,
                                                const float* __restrict__ a2,
                                                _Float16* __restrict__ whT,
                                                float* __restrict__ s1,
                                                float* __restrict__ s2) {
  __shared__ float sW[GAT_F * GAT_F];   // W[k][o]
  __shared__ float sT[64 * 72];         // h tile, then reused for Wh tile
  __shared__ float sa1[GAT_F], sa2[GAT_F];

  const int t  = threadIdx.x;
  const int R0 = blockIdx.x * 64;       // first flat row (b*N + n)

  for (int idx = t; idx < GAT_F * GAT_F; idx += 256) sW[idx] = W[idx];
  for (int idx = t; idx < 64 * GAT_F; idx += 256) {
    const int rr = idx >> 6, kk = idx & 63;
    sT[rr * 72 + kk] = h[(size_t)(R0 + rr) * GAT_F + kk];
  }
  if (t < GAT_F) { sa1[t] = a1[t]; sa2[t] = a2[t]; }
  __syncthreads();

  const int r  = t >> 2;          // local row 0..63
  const int cq = (t & 3) << 4;    // output-column group of 16
  float acc[16];
#pragma unroll
  for (int o = 0; o < 16; ++o) acc[o] = 0.f;
#pragma unroll 4
  for (int k = 0; k < GAT_F; ++k) {
    const float hv = sT[r * 72 + k];
#pragma unroll
    for (int o = 0; o < 16; ++o) acc[o] += hv * sW[k * GAT_F + cq + o];
  }
  __syncthreads();

  // Wh tile back into LDS for the s1/s2 dot products
#pragma unroll
  for (int o = 0; o < 16; ++o) sT[r * 72 + cq + o] = acc[o];

  // transposed f16 store: whT[b][o][n]  (block stays within one batch)
  const int b  = R0 >> 11;
  const int nr = (R0 & (GAT_N - 1)) + r;
#pragma unroll
  for (int o = 0; o < 16; ++o)
    whT[((size_t)(b * GAT_F + cq + o)) * GAT_N + nr] = (_Float16)acc[o];
  __syncthreads();

  if (t < 64) {
    float d1 = 0.f, d2 = 0.f;
#pragma unroll 8
    for (int o = 0; o < GAT_F; ++o) {
      const float v = sT[t * 72 + o];
      d1 += v * sa1[o];
      d2 += v * sa2[o];
    }
    s1[R0 + t] = d1;
    s2[R0 + t] = d2;
  }
}

// ---------------------------------------------------------------------------
// Kernel 2: fused scores + online softmax + attn@Wh via v_wmma_f32_16x16x32_f16.
// Block = 64 threads = 2 waves; each block owns one (batch, 16-row i-tile);
// the two waves split the j-range (1024 cols each) and merge through LDS.
//
// Per-wave layouts (wave32, per CDNA5 ISA):
//   A (P, 16x32 f16): lane = 16g+m holds row m, K in {8g..8g+7, 16+8g..16+8g+7}
//   B (Wh tile, 32x64 f16 as 4 o-tiles of 32x16): lane holds column n=m,
//       K in {8g..8g+7, 16+8g..16+8g+7} -> two 8-half runs of WhT[b][o0+m] at j0+8g and j0+16+8g
//   C/D (16x16 f32): VGPR v at lane 16g+m is element (M = v+8g, N = m)
// ---------------------------------------------------------------------------
__global__ __launch_bounds__(64) void gat_flash(const float* __restrict__ bias,
                                                const _Float16* __restrict__ whT,
                                                const float* __restrict__ s1,
                                                const float* __restrict__ s2,
                                                float* __restrict__ out) {
  __shared__ float smax[2][16], ssum[2][16];
  __shared__ float sacc[2][16][GAT_F];

  const int b    = blockIdx.x >> 7;              // 128 i-tiles per batch
  const int i0   = (blockIdx.x & 127) << 4;
  const int wave = threadIdx.x >> 5;
  const int lane = threadIdx.x & 31;
  const int g    = lane >> 4;
  const int m    = lane & 15;
  const int i    = i0 + m;

  const float     s1i      = s1[b * GAT_N + i];
  const float*    biasRow  = bias + ((size_t)b * GAT_N + i) * GAT_N;
  const float*    s2b      = s2 + b * GAT_N;
  const _Float16* whTb     = whT + (size_t)b * GAT_F * GAT_N;

  float mrun = -__builtin_inff();
  float lrun = 0.f;
  v8f acc[4];
#pragma unroll
  for (int ot = 0; ot < 4; ++ot)
#pragma unroll
    for (int v = 0; v < 8; ++v) acc[ot][v] = 0.f;

  const int jbeg = wave << 10;
  const int jend = jbeg + (GAT_N >> 1);

  for (int j0 = jbeg; j0 < jend; j0 += 32) {
    const int jA = j0 + (g << 3);   // this lane's first 8 K-columns
    const int jB = jA + 16;         // second 8

    // s2 slices (hot in L2) and bias slices (one-shot stream: non-temporal)
    const f32x4 sA0 = *(const f32x4*)(s2b + jA);
    const f32x4 sA1 = *(const f32x4*)(s2b + jA + 4);
    const f32x4 sB0 = *(const f32x4*)(s2b + jB);
    const f32x4 sB1 = *(const f32x4*)(s2b + jB + 4);
    const f32x4 bA0 = __builtin_nontemporal_load((const f32x4*)(biasRow + jA));
    const f32x4 bA1 = __builtin_nontemporal_load((const f32x4*)(biasRow + jA) + 1);
    const f32x4 bB0 = __builtin_nontemporal_load((const f32x4*)(biasRow + jB));
    const f32x4 bB1 = __builtin_nontemporal_load((const f32x4*)(biasRow + jB) + 1);
    if (j0 + 32 < jend) {           // prefetch next j-tile's bias stream
      __builtin_prefetch(biasRow + jA + 32, 0, 0);
      __builtin_prefetch(biasRow + jB + 32, 0, 0);
    }

    // e = leaky_relu(s1[i] + s2[j]) + bias[i][j]
    float e[16];
#pragma unroll
    for (int q = 0; q < 4; ++q) {
      float t0 = s1i + sA0[q]; t0 = t0 > 0.f ? t0 : GAT_SLOPE * t0; e[q]      = t0 + bA0[q];
      float t1 = s1i + sA1[q]; t1 = t1 > 0.f ? t1 : GAT_SLOPE * t1; e[q + 4]  = t1 + bA1[q];
      float t2 = s1i + sB0[q]; t2 = t2 > 0.f ? t2 : GAT_SLOPE * t2; e[q + 8]  = t2 + bB0[q];
      float t3 = s1i + sB1[q]; t3 = t3 > 0.f ? t3 : GAT_SLOPE * t3; e[q + 12] = t3 + bB1[q];
    }

    // online softmax: row spans lane m (g=0 half) and lane m+16 (g=1 half)
    float tmax = e[0];
#pragma unroll
    for (int q = 1; q < 16; ++q) tmax = fmaxf(tmax, e[q]);
    tmax = fmaxf(tmax, __shfl_xor(tmax, 16, 32));
    const float mnew  = fmaxf(mrun, tmax);
    const float alpha = __expf(mrun - mnew);   // first iter: exp(-inf)=0

    float p[16], tsum = 0.f;
#pragma unroll
    for (int q = 0; q < 16; ++q) { p[q] = __expf(e[q] - mnew); tsum += p[q]; }
    tsum += __shfl_xor(tsum, 16, 32);
    lrun = lrun * alpha + tsum;
    mrun = mnew;

    // rescale accumulators: acc row M = v + 8g needs alpha from lane (v+8g)
#pragma unroll
    for (int v = 0; v < 8; ++v) {
      const float av = __shfl(alpha, v + (g << 3), 32);
#pragma unroll
      for (int ot = 0; ot < 4; ++ot) acc[ot][v] *= av;
    }

    // pack P into the A-matrix layout (element order == p[0..15])
    v16h amat;
#pragma unroll
    for (int q = 0; q < 16; ++q) amat[q] = (_Float16)p[q];

    // 4 WMMAs: acc[ot] += P(16x32) x Wh(32x16) for o-tile ot
    const int jcol = j0 + (g << 3);
#pragma unroll
    for (int ot = 0; ot < 4; ++ot) {
      const _Float16* bp = whTb + (size_t)((ot << 4) + m) * GAT_N + jcol;
      v16h bmat;
#pragma unroll
      for (int s = 0; s < 16; ++s) bmat[s] = bp[s < 8 ? s : s + 8];
      acc[ot] = __builtin_amdgcn_wmma_f32_16x16x32_f16(
          /*neg_a=*/false, amat, /*neg_b=*/false, bmat,
          /*c_mod=*/(short)0, acc[ot], /*reuse_a=*/false, /*reuse_b=*/false);
    }
  }

  // merge the two waves' partial flash states through LDS
  if (g == 0) { smax[wave][m] = mrun; ssum[wave][m] = lrun; }
#pragma unroll
  for (int ot = 0; ot < 4; ++ot)
#pragma unroll
    for (int v = 0; v < 8; ++v)
      sacc[wave][(g << 3) + v][(ot << 4) + m] = acc[ot][v];
  __syncthreads();

  {
    const int r  = threadIdx.x >> 2;          // row of the i-tile
    const int c0 = (threadIdx.x & 3) << 4;    // 16-wide column group
    const float m0 = smax[0][r], m1 = smax[1][r];
    const float mm = fmaxf(m0, m1);
    const float w0 = __expf(m0 - mm), w1 = __expf(m1 - mm);
    const float inv = 1.0f / (ssum[0][r] * w0 + ssum[1][r] * w1);
    float* orow = out + ((size_t)b * GAT_N + i0 + r) * GAT_F + c0;
#pragma unroll
    for (int c = 0; c < 16; ++c)
      orow[c] = (sacc[0][r][c0 + c] * w0 + sacc[1][r][c0 + c] * w1) * inv;
  }
}

// ---------------------------------------------------------------------------
extern "C" void kernel_launch(void* const* d_in, const int* in_sizes, int n_in,
                              void* d_out, int out_size, void* d_ws, size_t ws_size,
                              hipStream_t stream) {
  const float* h    = (const float*)d_in[0];   // [16,2048,64]
  const float* bias = (const float*)d_in[1];   // [16,2048,2048]
  const float* W    = (const float*)d_in[2];   // [64,64]
  const float* a1   = (const float*)d_in[3];   // [64]
  const float* a2   = (const float*)d_in[4];   // [64]
  float* out = (float*)d_out;                  // [16,2048,64]

  char* ws = (char*)d_ws;
  _Float16* whT = (_Float16*)ws;                                   // 4 MiB
  float*    s1  = (float*)(ws + (size_t)4 * 1024 * 1024);          // 128 KiB
  float*    s2  = (float*)(ws + (size_t)4 * 1024 * 1024 + 131072); // 128 KiB

  gat_prep<<<(GAT_B * GAT_N) / 64, 256, 0, stream>>>(h, W, a1, a2, whT, s1, s2);
  gat_flash<<<GAT_B * (GAT_N / 16), 64, 0, stream>>>(bias, whT, s1, s2, out);
}
